// ArgreementRouting_37976100831587
// MI455X (gfx1250) — matrix operand order repaired and checked by reading, not compiled, not run on hardware
//
#include <hip/hip_runtime.h>
#include <hip/hip_bf16.h>
#include <math.h>

// ---------------- CDNA5 WMMA types ----------------
typedef __bf16 v16bf __attribute__((ext_vector_type(16)));
typedef float  v8f   __attribute__((ext_vector_type(8)));
typedef unsigned int u32x8 __attribute__((ext_vector_type(8)));

// ---------------- problem constants ----------------
#define IN_CAPS   32
#define IN_DIMS   288
#define OUT_CAPS  10
#define OUT_DIMS  16
#define BATCH     8192
#define NCOL      160      // OUT_CAPS*OUT_DIMS
#define KCHUNKS   9        // 288 / 32
#define NTILES    10       // 160 / 16
#define BTILES    512      // 8192 / 16
#define UHAT_ROW  (IN_CAPS*NCOL)      // 5120 floats per batch element
#define NPAIR     (IN_CAPS*OUT_CAPS)  // 320 routing coefficients
#define RBLOCKS   ((BATCH*OUT_CAPS)/16)  // 5120 routing blocks

// ws layout (float indices)
#define WS_UHAT   0u
#define N_UHAT    41943040u                  // 8192*32*160
#define WS_WHI    (N_UHAT)                   // 737280 u32
#define N_WFRAG   737280u                    // 32*9*10*8*32
#define WS_WLO    (WS_WHI + N_WFRAG)
#define WS_B      (WS_WLO + N_WFRAG)
#define WS_C      (WS_B + 320u)
#define WS_APART  (WS_C + 320u)              // 5120*320 floats

// round-to-nearest-even f32 -> bf16 bits
__device__ __forceinline__ unsigned f2bf(float x) {
  unsigned u = __float_as_uint(x);
  return (u + 0x7FFFu + ((u >> 16) & 1u)) >> 16;
}

// load a bf16 fragment stored as 8 consecutive packed dwords (2x b128)
__device__ __forceinline__ v16bf load_frag8(const unsigned* __restrict__ p) {
  uint4 x0 = *(const uint4*)(p);
  uint4 x1 = *(const uint4*)(p + 4);
  u32x8 v = (u32x8){x0.x, x0.y, x0.z, x0.w, x1.x, x1.y, x1.z, x1.w};
  return __builtin_bit_cast(v16bf, v);
}

// A fragment: pair indices base..base+7 (2x b128 from LDS)
__device__ __forceinline__ v16bf load_afrag(const unsigned* __restrict__ p) {
  uint4 x0 = *(const uint4*)(p);
  uint4 x1 = *(const uint4*)(p + 4);
  u32x8 v = (u32x8){x0.x, x0.y, x0.z, x0.w, x1.x, x1.y, x1.z, x1.w};
  return __builtin_bit_cast(v16bf, v);
}

// ---------------------------------------------------------------------------
// Kernel S: split W into bf16 hi/lo, permuted into the CDNA5 B-matrix
// (32x16 bf16) fragment layout, [lane][j] order so the GEMM loads b128:
// idx = (((cap*9+kc)*10 + ntile)*32 + lane)*8 + j
// ---------------------------------------------------------------------------
__global__ __launch_bounds__(256) void wsplit_kernel(
    const float* __restrict__ W,
    unsigned* __restrict__ whi, unsigned* __restrict__ wlo) {
  unsigned tid  = blockIdx.x * 256u + threadIdx.x;   // < 737280
  unsigned j    = tid & 7u;
  unsigned lane = (tid >> 3) & 31u;
  unsigned q    = tid >> 8;
  unsigned t    = q % 10u;
  unsigned r    = q / 10u;
  unsigned kc   = r % 9u;
  unsigned cap  = r / 9u;
  unsigned half = lane >> 4;
  unsigned n    = t * 16u + (lane & 15u);
  unsigned k0   = kc * 32u + half * 16u + 2u * j;
  float w0 = W[(cap * 288u + k0) * 160u + n];
  float w1 = W[(cap * 288u + k0 + 1u) * 160u + n];
  unsigned h0 = f2bf(w0), h1 = f2bf(w1);
  float hf0 = __uint_as_float(h0 << 16), hf1 = __uint_as_float(h1 << 16);
  unsigned l0 = f2bf(w0 - hf0), l1 = f2bf(w1 - hf1);
  whi[tid] = h0 | (h1 << 16);
  wlo[tid] = l0 | (l1 << 16);
}

// ---------------------------------------------------------------------------
// Kernel G: u_hat GEMM with 3xBF16 fp32-emulated WMMA.
// Block = (batch tile of 16, capsule). 5 waves, TWO 16-col N tiles per wave:
// A fragments loaded once per chunk feed 6 WMMAs.
// ---------------------------------------------------------------------------
__global__ __launch_bounds__(160) void uhat_gemm_kernel(
    const float* __restrict__ data,
    const unsigned* __restrict__ whi, const unsigned* __restrict__ wlo,
    float* __restrict__ uhat) {
  // packed (K even, K odd) bf16 pairs; row stride 148 -> conflict-free, 16B aligned
  __shared__ __align__(16) unsigned Apk[2][16][148];
  const unsigned btile = blockIdx.x;
  const unsigned cap   = blockIdx.y;
  const unsigned tid   = threadIdx.x;

  // Phase 1: load 16x288 fp32 A tile, split to bf16 hi/lo pairs in LDS
  for (unsigned p = tid; p < 16u * 144u; p += 160u) {
    unsigned row = p / 144u, kp = p % 144u;
    size_t base = (size_t)(btile * 16u + row) * (IN_CAPS * IN_DIMS)
                + cap * IN_DIMS + 2u * kp;
    float d0 = data[base], d1 = data[base + 1];
    unsigned h0 = f2bf(d0), h1 = f2bf(d1);
    float hf0 = __uint_as_float(h0 << 16), hf1 = __uint_as_float(h1 << 16);
    unsigned l0 = f2bf(d0 - hf0), l1 = f2bf(d1 - hf1);
    Apk[0][row][kp] = h0 | (h1 << 16);
    Apk[1][row][kp] = l0 | (l1 << 16);
  }
  __syncthreads();

  const unsigned wave = tid >> 5;     // 0..4
  const unsigned t0   = wave * 2u;    // first of two N tiles
  const unsigned lane = tid & 31u;
  const unsigned half = lane >> 4;
  const unsigned row  = lane & 15u;

  v8f acc0 = (v8f){0.f, 0.f, 0.f, 0.f, 0.f, 0.f, 0.f, 0.f};
  v8f acc1 = acc0;

  for (unsigned kc = 0; kc < KCHUNKS; ++kc) {
    // A fragments (16-bit A 16x32 layout)
    unsigned base1 = kc * 16u + 8u * half;
    v16bf Ah = load_afrag(&Apk[0][row][0] + base1);
    v16bf Al = load_afrag(&Apk[1][row][0] + base1);
    // B fragments: 2x b128 each, per-lane contiguous (L2-resident)
    unsigned wb0 = (((cap * 9u + kc) * 10u + t0) * 32u + lane) * 8u;
    unsigned wb1 = wb0 + 256u;        // next N tile
    v16bf Bh0 = load_frag8(whi + wb0);
    v16bf Bl0 = load_frag8(wlo + wb0);
    v16bf Bh1 = load_frag8(whi + wb1);
    v16bf Bl1 = load_frag8(wlo + wb1);
    acc0 = __builtin_amdgcn_wmma_f32_16x16x32_bf16(false, Ah, false, Bh0,
                                                   (short)0, acc0, false, false);
    acc1 = __builtin_amdgcn_wmma_f32_16x16x32_bf16(false, Ah, false, Bh1,
                                                   (short)0, acc1, false, false);
    acc0 = __builtin_amdgcn_wmma_f32_16x16x32_bf16(false, Ah, false, Bl0,
                                                   (short)0, acc0, false, false);
    acc1 = __builtin_amdgcn_wmma_f32_16x16x32_bf16(false, Ah, false, Bl1,
                                                   (short)0, acc1, false, false);
    acc0 = __builtin_amdgcn_wmma_f32_16x16x32_bf16(false, Al, false, Bh0,
                                                   (short)0, acc0, false, false);
    acc1 = __builtin_amdgcn_wmma_f32_16x16x32_bf16(false, Al, false, Bh1,
                                                   (short)0, acc1, false, false);
  }

  // Store C (16x16 f32 C/D layout): VGPR j -> M = j + 8*half, N = lane&15
  unsigned ncol0 = t0 * 16u + row;
#pragma unroll
  for (int j = 0; j < 8; ++j) {
    unsigned m = (unsigned)j + 8u * half;
    size_t o = (size_t)(btile * 16u + m) * UHAT_ROW + cap * NCOL + ncol0;
    uhat[o] = acc0[j];
    uhat[o + 16u] = acc1[j];
  }
}

// ---------------------------------------------------------------------------
// Kernel R: one routing iteration. Thread (x=d, y -> (b,o) pair).
// Pass1: v = sum_i c[i,o]*u_hat ; Pass2: agreement norms -> per-block partials
// (deterministic: no float atomics anywhere).
// ---------------------------------------------------------------------------
__global__ __launch_bounds__(256) void routing_kernel(
    const float* __restrict__ uhat, const float* __restrict__ cvec,
    float* __restrict__ v_out, float* __restrict__ a_part) {
  __shared__ float c_loc[NPAIR];
  __shared__ float sq[IN_CAPS][16];   // per-(i, y-row) sqrt agreement
  const unsigned x = threadIdx.x;     // 0..15 = d
  const unsigned y = threadIdx.y;     // 0..15 = local pair
  const unsigned tid = y * 16u + x;
  for (unsigned t = tid; t < NPAIR; t += 256u) c_loc[t] = cvec[t];
  __syncthreads();

  const unsigned P = blockIdx.x * 16u + y;
  const unsigned b = P / 10u;
  const unsigned o = P % 10u;
  const size_t base = (size_t)b * UHAT_ROW + o * 16u + x;

  float vacc = 0.f;
#pragma unroll
  for (unsigned i = 0; i < IN_CAPS; ++i)
    vacc += c_loc[i * 10u + o] * uhat[base + i * (unsigned)NCOL];
  v_out[(size_t)b * NCOL + o * 16u + x] = vacc;

#pragma unroll 4
  for (unsigned i = 0; i < IN_CAPS; ++i) {
    float p = uhat[base + i * (unsigned)NCOL] * vacc;
    float s = p * p;
    s += __shfl_xor(s, 1);
    s += __shfl_xor(s, 2);
    s += __shfl_xor(s, 4);
    s += __shfl_xor(s, 8);            // sum over the 16 d-lanes
    if (x == 0) sq[i][y] = sqrtf(s);
  }
  __syncthreads();

  // fold the 16 local pairs into 320 (i,o) partials in a fixed order
  unsigned base_o = (blockIdx.x * 16u) % 10u;
  for (unsigned t = tid; t < NPAIR; t += 256u) {
    unsigned i = t / 10u, oo = t % 10u;
    float acc = 0.f;
#pragma unroll
    for (unsigned yy = 0; yy < 16u; ++yy)
      if (((base_o + yy) % 10u) == oo) acc += sq[i][yy];
    a_part[(size_t)blockIdx.x * NPAIR + t] = acc;
  }
}

// ---------------------------------------------------------------------------
// Kernel RU: deterministic reduction of per-block partials, b += mean
// ---------------------------------------------------------------------------
__global__ __launch_bounds__(256) void reduce_update_kernel(
    const float* __restrict__ a_part, float* __restrict__ bvec) {
  __shared__ float red[256];
  unsigned e = blockIdx.x;            // 0..319
  float s = 0.f;
  for (unsigned p = threadIdx.x; p < (unsigned)RBLOCKS; p += 256u)
    s += a_part[(size_t)p * NPAIR + e];
  red[threadIdx.x] = s;
  __syncthreads();
  for (unsigned st = 128; st > 0; st >>= 1) {
    if (threadIdx.x < st) red[threadIdx.x] += red[threadIdx.x + st];
    __syncthreads();
  }
  if (threadIdx.x == 0) bvec[e] += red[0] * (1.0f / (float)BATCH);
}

// ---------------------------------------------------------------------------
// Kernel C: softmax over input capsules (axis i) for each o
// ---------------------------------------------------------------------------
__global__ void softmax_kernel(const float* __restrict__ bvec,
                               float* __restrict__ cvec) {
  unsigned o = threadIdx.x;
  if (o >= OUT_CAPS) return;
  float m = -1e30f;
  for (int i = 0; i < IN_CAPS; ++i) m = fmaxf(m, bvec[i * 10 + o]);
  float e[IN_CAPS];
  float s = 0.f;
  for (int i = 0; i < IN_CAPS; ++i) {
    e[i] = expf(bvec[i * 10 + o] - m);
    s += e[i];
  }
  float inv = 1.f / s;
  for (int i = 0; i < IN_CAPS; ++i) cvec[i * 10 + o] = e[i] * inv;
}

// ---------------------------------------------------------------------------
// Kernel I: zero logits, uniform initial coupling (softmax of zeros)
// ---------------------------------------------------------------------------
__global__ void init_kernel(float* __restrict__ bvec, float* __restrict__ cvec) {
  unsigned t = threadIdx.x;
  if (t < NPAIR) {
    bvec[t] = 0.f;
    cvec[t] = 1.0f / (float)IN_CAPS;
  }
}

// ---------------------------------------------------------------------------
extern "C" void kernel_launch(void* const* d_in, const int* in_sizes, int n_in,
                              void* d_out, int out_size, void* d_ws,
                              size_t ws_size, hipStream_t stream) {
  const float* data = (const float*)d_in[0];
  const float* W    = (const float*)d_in[1];
  float* out = (float*)d_out;
  float* ws  = (float*)d_ws;

  float*    uhat   = ws + WS_UHAT;
  unsigned* whi    = (unsigned*)(ws + WS_WHI);
  unsigned* wlo    = (unsigned*)(ws + WS_WLO);
  float*    bvec   = ws + WS_B;
  float*    cvec   = ws + WS_C;
  float*    a_part = ws + WS_APART;

  wsplit_kernel<<<N_WFRAG / 256u, 256, 0, stream>>>(W, whi, wlo);
  init_kernel<<<1, 320, 0, stream>>>(bvec, cvec);
  uhat_gemm_kernel<<<dim3(BTILES, IN_CAPS), 160, 0, stream>>>(data, whi, wlo,
                                                              uhat);
  for (int it = 0; it < 3; ++it) {
    routing_kernel<<<RBLOCKS, dim3(16, 16), 0, stream>>>(uhat, cvec, out,
                                                         a_part);
    if (it < 2) {
      reduce_update_kernel<<<NPAIR, 256, 0, stream>>>(a_part, bvec);
      softmax_kernel<<<1, 32, 0, stream>>>(bvec, cvec);
    }
  }
}
